// AffineAndSoftmaxLayer_88742614270065
// MI455X (gfx1250) — matrix operand order repaired and checked by reading, not compiled, not run on hardware
//
#include <hip/hip_runtime.h>
#include <hip/hip_bf16.h>

// ---------------------------------------------------------------------------
// AffineAndSoftmax: out = softmax(X[4096,12544] @ W[12544,1000] + b)
// bf16 WMMA (v_wmma_f32_16x16x32_bf16), f32 accumulate, explicit double-
// buffered K loop (steady-state body fully regular; last pair peeled),
// fused bias + softmax epilogue via LDS + wave32 shuffles.
// ---------------------------------------------------------------------------

typedef __attribute__((ext_vector_type(16))) __bf16        v16bf;
typedef __attribute__((ext_vector_type(8)))  float         v8f;
typedef __attribute__((ext_vector_type(4)))  unsigned int  vu4;   // 16B, POD, b128-friendly

#define B_ROWS  4096
#define K_DIM   12544          // = 392 * 32
#define KT      392
#define D_OUT   1000
#define NT_PAD  64             // 64 n-tiles of 16 -> 1024 padded columns
#define N_PAD   (NT_PAD * 16)

struct Frag { vu4 u[2]; };     // 32 bytes = one 16-bit 16x32 (A) or 32x16 (B) fragment per lane

__device__ __forceinline__ v16bf frag_as_v16bf(const Frag& f) {
  union { Frag f; v16bf v; } c;
  c.f = f;
  return c.v;
}

__device__ __forceinline__ unsigned short f32_to_bf16(float f) {
  unsigned int u = __float_as_uint(f);
  unsigned int r = (u + 0x7FFFu + ((u >> 16) & 1u)) >> 16;  // RNE
  return (unsigned short)r;
}

// --- Kernel 1: X fp32 -> bf16 row-major, 8 elements/thread, b128 store ------
__global__ __launch_bounds__(256) void cvt_x_kernel(
    const float* __restrict__ x, unsigned short* __restrict__ y, long n) {
  long i = ((long)blockIdx.x * blockDim.x + threadIdx.x) * 8;
  if (i >= n) return;
  union { unsigned short s[8]; vu4 u; } t;
#pragma unroll
  for (int j = 0; j < 8; ++j) t.s[j] = f32_to_bf16(x[i + j]);
  *(vu4*)(y + i) = t.u;
}

// --- Kernel 2: W fp32 -> bf16, swizzled into WMMA B-fragment blocks ---------
// Layout: frag[((kt*NT_PAD + nt)*32 + lane)*16 .. +15]
//   lane L: n = nt*16 + (L&15), half = L>>4
//   element e (VGPR e/2, hi-half e&1): k = kt*32 + 16*half + e
// Zero-fill padded columns (n >= 1000).
__global__ __launch_bounds__(256) void cvt_w_kernel(
    const float* __restrict__ w, unsigned short* __restrict__ frag) {
  long t = (long)blockIdx.x * blockDim.x + threadIdx.x;
  if (t >= (long)KT * NT_PAD * 32) return;
  int  lane = (int)(t & 31);
  int  nt   = (int)((t >> 5) & (NT_PAD - 1));
  long kt   = t >> 11;                       // 32*64 = 2048 = 2^11
  int  half = lane >> 4;
  int  n    = nt * 16 + (lane & 15);
  union { unsigned short s[16]; vu4 u[2]; } o;
#pragma unroll
  for (int e = 0; e < 16; ++e) {
    int k = (int)kt * 32 + 16 * half + e;
    float v = (n < D_OUT) ? w[(size_t)k * D_OUT + n] : 0.0f;
    o.s[e] = f32_to_bf16(v);
  }
  vu4* dst = (vu4*)(frag + t * 16);
  dst[0] = o.u[0];
  dst[1] = o.u[1];
}

// --- fragment loads ---------------------------------------------------------
// A in contiguous k order (matches B): lanes 0-15: K 0..15; lanes 16-31: K 16..31
// -> two b128 loads at k offsets half*16, half*16+8.
__device__ __forceinline__ void load_a(const unsigned short* __restrict__ Arow,
                                       int kt, Frag& a) {
  const vu4* ap = (const vu4*)(Arow + (size_t)kt * 32);
  a.u[0] = ap[0];   // 16B @ k = half*16
  a.u[1] = ap[1];   // 16B @ k = half*16 + 8
}

__device__ __forceinline__ void load_b4(const unsigned short* __restrict__ Wbase,
                                        int kt, Frag b[4]) {
  const unsigned short* wk = Wbase + (size_t)kt * (NT_PAD * 32 * 16);
#pragma unroll
  for (int i = 0; i < 4; ++i) {
    const vu4* bp = (const vu4*)(wk + (size_t)i * (32 * 16));
    b[i].u[0] = bp[0];
    b[i].u[1] = bp[1];
  }
}

__device__ __forceinline__ void wmma4(const Frag& a, const Frag b[4], v8f acc[4]) {
#pragma unroll
  for (int i = 0; i < 4; ++i)
    acc[i] = __builtin_amdgcn_wmma_f32_16x16x32_bf16(
        false, frag_as_v16bf(a), false, frag_as_v16bf(b[i]),
        (short)0, acc[i], false, false);
}

// --- Kernel 3: fused GEMM + bias + softmax ----------------------------------
// grid.x = 256 (M tiles of 16 rows); block = 512 threads = 16 wave32s.
// Each wave owns 4 n-tiles -> 16*4 = 64 = NT_PAD tiles, branch-free K loop.
__global__ __launch_bounds__(512, 1) void gemm_bias_softmax_kernel(
    const unsigned short* __restrict__ Xbf,    // [4096][12544] bf16
    const unsigned short* __restrict__ Wfrag,  // fragment blocks (see cvt_w)
    const float* __restrict__ bias,            // [1000]
    float* __restrict__ out)                   // [4096][1000]
{
  __shared__ float smem[16 * N_PAD];           // 64 KB: exp(logits) slab

  const int tid   = threadIdx.x;
  const int wave  = tid >> 5;
  const int lane  = tid & 31;
  const int half  = lane >> 4;
  const int l16   = lane & 15;
  const int mtile = blockIdx.x;
  const int nt0   = wave * 4;

  v8f acc[4];
#pragma unroll
  for (int i = 0; i < 4; ++i) acc[i] = v8f{};

  const unsigned short* Arow  = Xbf + (size_t)(mtile * 16 + l16) * K_DIM + half * 16;
  const unsigned short* Wbase = Wfrag + ((size_t)nt0 * 32 + lane) * 16;

  // Explicit double buffering with two disjoint register sets. Steady-state
  // loop body is fully regular (affine addresses, no selects) so each WMMA
  // waits only on loads issued one iteration earlier; final pair is peeled.
  Frag a[2];
  Frag b[2][4];
  load_a(Arow, 0, a[0]);
  load_b4(Wbase, 0, b[0]);

  for (int kt = 0; kt < KT - 2; kt += 2) {           // kt = 0 .. 388
    load_a(Arow, kt + 1, a[1]);
    load_b4(Wbase, kt + 1, b[1]);
    wmma4(a[0], b[0], acc);

    load_a(Arow, kt + 2, a[0]);
    load_b4(Wbase, kt + 2, b[0]);
    wmma4(a[1], b[1], acc);
  }
  // Peeled tail: set0 holds kt = KT-2; prefetch KT-1, no bogus loads.
  load_a(Arow, KT - 1, a[1]);
  load_b4(Wbase, KT - 1, b[1]);
  wmma4(a[0], b[0], acc);
  wmma4(a[1], b[1], acc);

  // Epilogue: bias + exp -> LDS. C/D layout: VGPR r, lanes 0-15 -> (M=r, N=lane),
  // lanes 16-31 -> (M=8+r, N=lane-16).
#pragma unroll
  for (int i = 0; i < 4; ++i) {
    int col = (nt0 + i) * 16 + l16;
    float bv = (col < D_OUT) ? bias[col] : 0.0f;
#pragma unroll
    for (int r = 0; r < 8; ++r) {
      int mrow = half * 8 + r;
      float e = (col < D_OUT) ? __expf(acc[i][r] + bv) : 0.0f;
      smem[mrow * N_PAD + col] = e;
    }
  }
  __syncthreads();

  // Wave w reduces & writes row w. 32 lanes stride the 1024 columns (bank-clean).
  const int row = wave;
  float part = 0.0f;
#pragma unroll
  for (int j = 0; j < 32; ++j) part += smem[row * N_PAD + lane + 32 * j];
#pragma unroll
  for (int off = 16; off >= 1; off >>= 1) part += __shfl_xor(part, off, 32);
  float inv = 1.0f / part;

  size_t obase = (size_t)(mtile * 16 + row) * D_OUT;
#pragma unroll
  for (int j = 0; j < 32; ++j) {
    int col = lane + 32 * j;
    if (col < D_OUT) out[obase + col] = smem[row * N_PAD + col] * inv;
  }
}

// ---------------------------------------------------------------------------
extern "C" void kernel_launch(void* const* d_in, const int* in_sizes, int n_in,
                              void* d_out, int out_size, void* d_ws, size_t ws_size,
                              hipStream_t stream) {
  const float* inp  = (const float*)d_in[0];   // [4096,64,14,14] fp32 == [4096,12544]
  const float* w    = (const float*)d_in[1];   // [12544,1000] fp32
  const float* bias = (const float*)d_in[2];   // [1000] fp32
  float* out = (float*)d_out;

  unsigned short* wsX = (unsigned short*)d_ws;                 // 102,760,448 B
  size_t xBytes = (size_t)B_ROWS * K_DIM * sizeof(unsigned short);
  unsigned short* wsW = (unsigned short*)((char*)d_ws + xBytes); // +25,690,112 B

  long nX = (long)B_ROWS * K_DIM;                              // 51,380,224 (÷8 exact)
  cvt_x_kernel<<<(unsigned)(nX / 8 / 256), 256, 0, stream>>>(inp, wsX, nX);

  long nW = (long)KT * NT_PAD * 32;                            // 802,816
  cvt_w_kernel<<<(unsigned)((nW + 255) / 256), 256, 0, stream>>>(w, wsW);

  gemm_bias_softmax_kernel<<<B_ROWS / 16, 512, 0, stream>>>(wsX, wsW, bias, out);
}
